// CustomGRUCell_16758962389414
// MI455X (gfx1250) — compile-verified, matrix operand order checked by reading, not run on hardware
//
#include <hip/hip_runtime.h>

// ---------------------------------------------------------------------------
// GRU cell on MI455X (gfx1250, wave32, WMMA).
// fp32 -> bf16 cast, then 3 fused dual-GEMM kernels (one per gate) using
// v_wmma_f32_16x16x32_bf16 with f32 accumulation. 64x64 wave tiles
// (16 wmma : 16 global_load_b128 per K-step), 256x128 block tiles.
// All global accesses forced to address_space(1) => global_load/global_store.
// ---------------------------------------------------------------------------

typedef __attribute__((ext_vector_type(16))) __bf16          v16bf;
typedef __attribute__((ext_vector_type(8)))  float           v8f;
typedef __attribute__((ext_vector_type(4)))  unsigned int    v4u;
typedef __attribute__((ext_vector_type(4)))  float           v4f;
typedef __attribute__((ext_vector_type(4)))  unsigned short  v4us;

// Global (addrspace 1) pointer typedefs — force global_load/store codegen.
typedef __attribute__((address_space(1))) const v4u*           gc_v4u_p;
typedef __attribute__((address_space(1))) const float*         gc_f32_p;
typedef __attribute__((address_space(1))) float*               g_f32_p;
typedef __attribute__((address_space(1))) const v4f*           gc_v4f_p;
typedef __attribute__((address_space(1))) v4us*                g_v4us_p;
typedef __attribute__((address_space(1))) unsigned short*      g_u16_p;

#define B_DIM 8192
#define K_DIM 2048   // IN == H == 2048, so all row strides are identical
#define H_DIM 2048

// fp32 -> bf16 round-to-nearest-even
__device__ __forceinline__ unsigned short f2bf(float f) {
    unsigned int u = __builtin_bit_cast(unsigned int, f);
    u += 0x7FFFu + ((u >> 16) & 1u);
    return (unsigned short)(u >> 16);
}

__global__ void f32_to_bf16_kernel(const float* __restrict__ src,
                                   unsigned short* __restrict__ dst, int n) {
    int i = (blockIdx.x * blockDim.x + threadIdx.x) * 4;
    if (i >= n) return;
    v4f f = *(gc_v4f_p)(src + i);
    v4us o;
    o.x = f2bf(f.x);
    o.y = f2bf(f.y);
    o.z = f2bf(f.z);
    o.w = f2bf(f.w);
    *(g_v4us_p)(dst + i) = o;
}

union FragCast {
    v4u   u[2];
    v16bf bf;
};

// Load one 16x32 (A) or 32x16 (B) bf16 fragment for wave32 WMMA.
// Per the CDNA5 ISA 16-bit fragment layout, lane L (L&15 selects row/col,
// L>>4 selects K-half) needs K-chunks [ksel..ksel+7] and [16+ksel..16+ksel+7],
// both contiguous in a K-major source => two global_load_b128.
__device__ __forceinline__ v16bf load_frag(const unsigned short* __restrict__ base,
                                           int row0, int kb, int lane) {
    const int lrow = lane & 15;
    const int ksel = (lane >> 4) << 3;  // 0 or 8
    gc_v4u_p g = (gc_v4u_p)(base + (size_t)(row0 + lrow) * K_DIM + kb + ksel);
    FragCast f;
    f.u[0] = g[0];   // K = kb+ksel    .. +7
    f.u[1] = g[2];   // K = kb+16+ksel .. +7  (+32 bytes)
    return f.bf;
}

// MODE 0: r-gate  (sigmoid; also emit bf16(r*h) for the n-gate GEMM)
// MODE 1: z-gate  (sigmoid)
// MODE 2: n-gate  (tanh; also emit h_t = (1-z)*n + z*h)
template <int MODE>
__global__ __launch_bounds__(256)
void gru_gemm_kernel(const unsigned short* __restrict__ A0,   // bf16 [B, 2048]
                     const unsigned short* __restrict__ W0,   // bf16 [H, 2048]
                     const unsigned short* __restrict__ A1,   // bf16 [B, 2048]
                     const unsigned short* __restrict__ W1,   // bf16 [H, 2048]
                     const float* __restrict__ bias,          // fp32 [H]
                     const float* __restrict__ h,             // fp32 [B, H]
                     float* __restrict__ out_act,             // r / z / n
                     float* __restrict__ out_ht,              // MODE2: h_t
                     const float* __restrict__ z_in,          // MODE2: z
                     unsigned short* __restrict__ rh_out)     // MODE0: bf16(r*h)
{
    const int lane   = threadIdx.x & 31;
    const int wave   = threadIdx.x >> 5;
    const int wave_m = wave >> 1;   // 0..3
    const int wave_n = wave & 1;    // 0..1
    const int blockN = blockIdx.x * 128 + wave_n * 64;  // hidden-unit dim
    const int blockM = blockIdx.y * 256 + wave_m * 64;  // batch dim

    v8f acc[4][4];
#pragma unroll
    for (int i = 0; i < 4; ++i)
#pragma unroll
        for (int j = 0; j < 4; ++j)
            acc[i][j] = {};

    const unsigned short* Aptr[2] = {A0, A1};
    const unsigned short* Wptr[2] = {W0, W1};

#pragma unroll 1
    for (int g = 0; g < 2; ++g) {
        const unsigned short* __restrict__ A = Aptr[g];
        const unsigned short* __restrict__ W = Wptr[g];
#pragma unroll 2
        for (int kb = 0; kb < K_DIM; kb += 32) {
            v16bf a[4], b[4];
#pragma unroll
            for (int i = 0; i < 4; ++i)
                a[i] = load_frag(A, blockM + i * 16, kb, lane);
#pragma unroll
            for (int j = 0; j < 4; ++j)
                b[j] = load_frag(W, blockN + j * 16, kb, lane);

#pragma unroll
            for (int i = 0; i < 4; ++i)
#pragma unroll
                for (int j = 0; j < 4; ++j)
                    acc[i][j] = __builtin_amdgcn_wmma_f32_16x16x32_bf16(
                        false, a[i], false, b[j], (short)0, acc[i][j], false, false);
        }
    }

    // Epilogue. C layout (32-bit, 16x16): lane = n + 16*(m>=8), vgpr e => m = e + 8*(lane>>4).
    const int col0   = lane & 15;
    const int rowoff = (lane >> 4) * 8;
#pragma unroll
    for (int i = 0; i < 4; ++i) {
#pragma unroll
        for (int j = 0; j < 4; ++j) {
            const int n    = blockN + j * 16 + col0;
            const float bv = ((gc_f32_p)bias)[n];
#pragma unroll
            for (int e = 0; e < 8; ++e) {
                const int m = blockM + i * 16 + rowoff + e;
                const size_t idx = (size_t)m * H_DIM + n;
                const float v = acc[i][j][e] + bv;
                if (MODE == 2) {
                    const float nv = tanhf(v);
                    const float zv = ((gc_f32_p)z_in)[idx];
                    const float hv = ((gc_f32_p)h)[idx];
                    ((g_f32_p)out_act)[idx] = nv;
                    ((g_f32_p)out_ht)[idx]  = (1.0f - zv) * nv + zv * hv;
                } else {
                    const float s = 1.0f / (1.0f + __expf(-v));
                    ((g_f32_p)out_act)[idx] = s;
                    if (MODE == 0) {
                        ((g_u16_p)rh_out)[idx] = f2bf(s * ((gc_f32_p)h)[idx]);
                    }
                }
            }
        }
    }
}

extern "C" void kernel_launch(void* const* d_in, const int* in_sizes, int n_in,
                              void* d_out, int out_size, void* d_ws, size_t ws_size,
                              hipStream_t stream) {
    (void)in_sizes; (void)n_in; (void)out_size; (void)ws_size;

    const float* x    = (const float*)d_in[0];
    const float* h    = (const float*)d_in[1];
    const float* W_ir = (const float*)d_in[2];
    const float* b_ir = (const float*)d_in[3];
    const float* W_hr = (const float*)d_in[4];
    const float* W_iz = (const float*)d_in[5];
    const float* b_iz = (const float*)d_in[6];
    const float* W_hz = (const float*)d_in[7];
    const float* W_in = (const float*)d_in[8];
    const float* b_in = (const float*)d_in[9];
    const float* W_hn = (const float*)d_in[10];

    const size_t NBH = (size_t)B_DIM * H_DIM;      // 16M elements
    const size_t NW  = (size_t)H_DIM * K_DIM;      // 4M elements

    // Workspace layout (bf16), total 144 MB.
    unsigned short* x_bf  = (unsigned short*)d_ws;
    unsigned short* h_bf  = x_bf  + NBH;
    unsigned short* rh_bf = h_bf  + NBH;
    unsigned short* wir   = rh_bf + NBH;
    unsigned short* whr   = wir + NW;
    unsigned short* wiz   = whr + NW;
    unsigned short* whz   = wiz + NW;
    unsigned short* win   = whz + NW;
    unsigned short* whn   = win + NW;

    auto conv = [&](const float* s, unsigned short* d, size_t n) {
        const int threads = 256;
        const int blocks  = (int)((n / 4 + threads - 1) / threads);
        f32_to_bf16_kernel<<<blocks, threads, 0, stream>>>(s, d, (int)n);
    };
    conv(x,    x_bf, NBH);
    conv(h,    h_bf, NBH);
    conv(W_ir, wir,  NW);
    conv(W_hr, whr,  NW);
    conv(W_iz, wiz,  NW);
    conv(W_hz, whz,  NW);
    conv(W_in, win,  NW);
    conv(W_hn, whn,  NW);

    float* out    = (float*)d_out;
    float* out_ht = out;             // return order: h_t, r, z, n
    float* out_r  = out + NBH;
    float* out_z  = out + 2 * NBH;
    float* out_n  = out + 3 * NBH;

    dim3 grid(H_DIM / 128, B_DIM / 256);  // 16 x 32 blocks, 256 threads each
    // r-gate: sigmoid(x@W_ir^T + b_ir + h@W_hr^T); emit bf16(r*h)
    gru_gemm_kernel<0><<<grid, 256, 0, stream>>>(x_bf, wir, h_bf, whr, b_ir, h,
                                                 out_r, nullptr, nullptr, rh_bf);
    // z-gate: sigmoid(x@W_iz^T + b_iz + h@W_hz^T)
    gru_gemm_kernel<1><<<grid, 256, 0, stream>>>(x_bf, wiz, h_bf, whz, b_iz, h,
                                                 out_z, nullptr, nullptr, nullptr);
    // n-gate: tanh(x@W_in^T + b_in + (r*h)@W_hn^T); h_t = (1-z)*n + z*h
    gru_gemm_kernel<2><<<grid, 256, 0, stream>>>(x_bf, win, rh_bf, whn, b_in, h,
                                                 out_n, out_ht, out_z, nullptr);
}
